// Event_Temporal_28243704939270
// MI455X (gfx1250) — matrix operand order repaired and checked by reading, not compiled, not run on hardware
//
#include <hip/hip_runtime.h>
#include <math.h>

// ---------------------------------------------------------------------------
// Types for CDNA5 WMMA (wave32)
// ---------------------------------------------------------------------------
typedef __attribute__((ext_vector_type(16))) __bf16 v16bf;
typedef __attribute__((ext_vector_type(4)))  __bf16 v4bf;
typedef __attribute__((ext_vector_type(8)))  float  v8f;

#define L_SEQ 4096
#define D_IN  768
#define H_DIM 384
#define G4H   1536   // 4*H
#define NGATE 3072   // both directions stacked
#define E_NUM 32768
#define P_NUM 32768
#define KFEAT 2304   // [e1 | e2 | e1*e2]

// ---------------------------------------------------------------------------
// Fragment loads from a running pointer (two b128 loads, immediate offsets).
// A-frag (16x32): lane<16 -> row M=lane,   K {0..7,16..23}
//                 lane>=16 -> row M=lane-16, K {8..15,24..31}
// B-frag (32x16): lane<16 -> col N=lane,   K {0..7,16..23}
//                 lane>=16 -> col N=lane-16, K {8..15,24..31}  (W is row-major (N,K))
// ---------------------------------------------------------------------------
__device__ __forceinline__ v16bf ld_frag_a(const __bf16* p) {
    union { uint4 u[2]; v16bf v; } u;
    u.u[0] = *(const uint4*)(p);
    u.u[1] = *(const uint4*)(p + 16);
    return u.v;
}
__device__ __forceinline__ v16bf ld_frag_b(const __bf16* p) {
    union { uint4 u[2]; v16bf v; } u;
    u.u[0] = *(const uint4*)(p);
    u.u[1] = *(const uint4*)(p + 16);
    return u.v;
}

// ---------------------------------------------------------------------------
// Register-blocked bf16 WMMA GEMM: each wave computes a 32(M) x 64(N) tile.
// Six induction pointers (+32 elems/step) keep address pressure minimal;
// fragments for K-step k+32 are loaded before the 8 WMMAs of step k issue.
// C(M,N) = A(M,K) @ W(N,K)^T + bias, optional relu. 8 waves / block.
// Requires M % 32 == 0, N % 64 == 0, K % 32 == 0.
// ---------------------------------------------------------------------------
__global__ void __attribute__((amdgpu_waves_per_eu(2)))
gemm_bf16_wmma(const __bf16* __restrict__ A, int lda,
               const __bf16* __restrict__ W, int ldw,
               const float* __restrict__ bias,
               float* __restrict__ Cf, __bf16* __restrict__ Cb,
               int M, int N, int K, int relu)
{
    int lane   = threadIdx.x & 31;
    int wave   = threadIdx.x >> 5;
    int tilesN = N >> 6;
    int tilesM = M >> 5;
    int tid    = blockIdx.x * (blockDim.x >> 5) + wave;
    if (tid >= tilesM * tilesN) return;
    int tM = (tid / tilesN) << 5;
    int tN = (tid % tilesN) << 6;

    // induction pointers, one per fragment stream
    const __bf16* pa0 = A + (size_t)(tM + (lane & 15)) * lda + ((lane & 16) ? 8 : 0);
    const __bf16* pa1 = pa0 + (size_t)16 * lda;
    const __bf16* pb0 = W + (size_t)(tN + (lane & 15)) * ldw + ((lane & 16) ? 8 : 0);
    const __bf16* pb1 = pb0 + (size_t)16 * ldw;
    const __bf16* pb2 = pb0 + (size_t)32 * ldw;
    const __bf16* pb3 = pb0 + (size_t)48 * ldw;

    v8f acc[2][4] = {};

    // prologue: fragments for k = 0
    v16bf a0 = ld_frag_a(pa0);
    v16bf a1 = ld_frag_a(pa1);
    v16bf b0 = ld_frag_b(pb0);
    v16bf b1 = ld_frag_b(pb1);
    v16bf b2 = ld_frag_b(pb2);
    v16bf b3 = ld_frag_b(pb3);

    for (int k = 0; k + 32 < K; k += 32) {
        pa0 += 32; pa1 += 32; pb0 += 32; pb1 += 32; pb2 += 32; pb3 += 32;
        // speculative far-ahead prefetch (~7 K-steps); dropped if invalid
        __builtin_prefetch(pa0 + 224, 0, 0);
        __builtin_prefetch(pb0 + 224, 0, 0);

        // next K-step's fragments issue before current WMMAs consume
        v16bf na0 = ld_frag_a(pa0);
        v16bf na1 = ld_frag_a(pa1);
        v16bf nb0 = ld_frag_b(pb0);
        v16bf nb1 = ld_frag_b(pb1);
        v16bf nb2 = ld_frag_b(pb2);
        v16bf nb3 = ld_frag_b(pb3);

        acc[0][0] = __builtin_amdgcn_wmma_f32_16x16x32_bf16(
            false, a0, false, b0, (short)0, acc[0][0], false, false);
        acc[1][0] = __builtin_amdgcn_wmma_f32_16x16x32_bf16(
            false, a1, false, b0, (short)0, acc[1][0], false, false);
        acc[0][1] = __builtin_amdgcn_wmma_f32_16x16x32_bf16(
            false, a0, false, b1, (short)0, acc[0][1], false, false);
        acc[1][1] = __builtin_amdgcn_wmma_f32_16x16x32_bf16(
            false, a1, false, b1, (short)0, acc[1][1], false, false);
        acc[0][2] = __builtin_amdgcn_wmma_f32_16x16x32_bf16(
            false, a0, false, b2, (short)0, acc[0][2], false, false);
        acc[1][2] = __builtin_amdgcn_wmma_f32_16x16x32_bf16(
            false, a1, false, b2, (short)0, acc[1][2], false, false);
        acc[0][3] = __builtin_amdgcn_wmma_f32_16x16x32_bf16(
            false, a0, false, b3, (short)0, acc[0][3], false, false);
        acc[1][3] = __builtin_amdgcn_wmma_f32_16x16x32_bf16(
            false, a1, false, b3, (short)0, acc[1][3], false, false);

        a0 = na0; a1 = na1; b0 = nb0; b1 = nb1; b2 = nb2; b3 = nb3;
    }

    // epilogue K-step
    acc[0][0] = __builtin_amdgcn_wmma_f32_16x16x32_bf16(
        false, a0, false, b0, (short)0, acc[0][0], false, false);
    acc[1][0] = __builtin_amdgcn_wmma_f32_16x16x32_bf16(
        false, a1, false, b0, (short)0, acc[1][0], false, false);
    acc[0][1] = __builtin_amdgcn_wmma_f32_16x16x32_bf16(
        false, a0, false, b1, (short)0, acc[0][1], false, false);
    acc[1][1] = __builtin_amdgcn_wmma_f32_16x16x32_bf16(
        false, a1, false, b1, (short)0, acc[1][1], false, false);
    acc[0][2] = __builtin_amdgcn_wmma_f32_16x16x32_bf16(
        false, a0, false, b2, (short)0, acc[0][2], false, false);
    acc[1][2] = __builtin_amdgcn_wmma_f32_16x16x32_bf16(
        false, a1, false, b2, (short)0, acc[1][2], false, false);
    acc[0][3] = __builtin_amdgcn_wmma_f32_16x16x32_bf16(
        false, a0, false, b3, (short)0, acc[0][3], false, false);
    acc[1][3] = __builtin_amdgcn_wmma_f32_16x16x32_bf16(
        false, a1, false, b3, (short)0, acc[1][3], false, false);

    // C/D layout: lane<16 -> N=lane, M = v; lane>=16 -> N=lane-16, M = 8+v
    int nl   = lane & 15;
    int msel = (lane & 16) ? 8 : 0;
#pragma unroll
    for (int mi = 0; mi < 2; ++mi) {
#pragma unroll
        for (int j = 0; j < 4; ++j) {
            int ncol = tN + 16 * j + nl;
            float bv = bias ? bias[ncol] : 0.0f;
#pragma unroll
            for (int v = 0; v < 8; ++v) {
                float x = acc[mi][j][v] + bv;
                if (relu) x = fmaxf(x, 0.0f);
                size_t idx = (size_t)(tM + mi * 16 + msel + v) * N + ncol;
                if (Cf) Cf[idx] = x;
                if (Cb) Cb[idx] = (__bf16)x;
            }
        }
    }
}

// ---------------------------------------------------------------------------
// Prep kernels (f32 -> bf16 conversions / weight folding / transposes)
// ---------------------------------------------------------------------------
__global__ void k_f32_to_bf16(const float* __restrict__ s, __bf16* __restrict__ d, int n) {
    int i = blockIdx.x * blockDim.x + threadIdx.x;
    if (i < n) d[i] = (__bf16)s[i];
}

__global__ void k_build_wih(const float* __restrict__ wf, const float* __restrict__ wb,
                            __bf16* __restrict__ dst) {
    int i = blockIdx.x * blockDim.x + threadIdx.x;     // over 3072*768
    if (i >= NGATE * D_IN) return;
    int row = i / D_IN, k = i - row * D_IN;
    float v = (row < G4H) ? wf[row * D_IN + k] : wb[(row - G4H) * D_IN + k];
    dst[i] = (__bf16)v;
}

__global__ void k_build_bias(const float* __restrict__ bf_, const float* __restrict__ bb_,
                             float* __restrict__ dst) {
    int i = blockIdx.x * blockDim.x + threadIdx.x;     // over 3072
    if (i >= NGATE) return;
    dst[i] = (i < G4H) ? bf_[i] : bb_[i - G4H];
}

// Transpose (4H,H) -> k-major, interleaved in groups of 4 k's:
// dst[((k>>2)*G4H + j)*4 + (k&3)] = W[j][k]  -> 8B per lane in the matvec.
__global__ void k_build_whhT(const float* __restrict__ whf, const float* __restrict__ whb,
                             __bf16* __restrict__ dst) {
    int i = blockIdx.x * blockDim.x + threadIdx.x;     // over 2*384*1536
    if (i >= 2 * H_DIM * G4H) return;
    int dir = i / (H_DIM * G4H);
    int r   = i - dir * (H_DIM * G4H);
    int k   = r / G4H;          // 0..383
    int j   = r - k * G4H;      // 0..1535
    const float* src = dir ? whb : whf;
    size_t base = (size_t)dir * H_DIM * G4H;
    dst[base + ((size_t)(k >> 2) * G4H + j) * 4 + (k & 3)] =
        (__bf16)src[(size_t)j * H_DIM + k];
}

// Fold W1 (768,3072) into (768,2304): [W1a+W1c | W1b-W1c | W1d]
__global__ void k_build_wmlp1(const float* __restrict__ w1, __bf16* __restrict__ dst) {
    int i = blockIdx.x * blockDim.x + threadIdx.x;     // over 768*2304
    if (i >= D_IN * KFEAT) return;
    int n = i / KFEAT, kp = i - n * KFEAT;
    const float* row = w1 + (size_t)n * 3072;
    float v;
    if (kp < 768)        v = row[kp] + row[1536 + kp];
    else if (kp < 1536)  { int d = kp - 768;  v = row[768 + d] - row[1536 + d]; }
    else                 { int d = kp - 1536; v = row[2304 + d]; }
    dst[i] = (__bf16)v;
}

// ---------------------------------------------------------------------------
// LSTM recurrence: one workgroup per direction, h/c in LDS.
// Gate matvec reads 4-way k-packed bf16 W_hh^T (8B/lane, coalesced along j).
// ---------------------------------------------------------------------------
__global__ void __launch_bounds__(1024)
lstm_kernel(const float* __restrict__ Gx,        // (L, 3072) x@Wih^T + b, both dirs
            const __bf16* __restrict__ WhhT,     // (2, 96, 1536, 4) packed
            float* __restrict__ tok)             // (L, 768) [fwd | bwd]
{
    __shared__ float h_s[H_DIM];
    __shared__ float c_s[H_DIM];
    __shared__ float gates[G4H];

    int dir = blockIdx.x;
    int tid = threadIdx.x;
    const v4bf* Wd = (const v4bf*)(WhhT + (size_t)dir * H_DIM * G4H);

    if (tid < H_DIM) { h_s[tid] = 0.0f; c_s[tid] = 0.0f; }
    __syncthreads();

    for (int s = 0; s < L_SEQ; ++s) {
        int t = dir ? (L_SEQ - 1 - s) : s;
        const float* g0 = Gx + (size_t)t * NGATE + dir * G4H;

        // prefetch next step's gate row while we compute this one
        if (s + 1 < L_SEQ) {
            int tn = dir ? (L_SEQ - 2 - s) : (s + 1);
            const float* gn = Gx + (size_t)tn * NGATE + dir * G4H;
            if (tid < G4H) __builtin_prefetch(gn + tid, 0, 0);
        }

        for (int j = tid; j < G4H; j += 1024) {
            float acc = g0[j];
            const v4bf* wcol = Wd + j;
#pragma unroll 2
            for (int k4 = 0; k4 < H_DIM / 4; ++k4) {
                v4bf w = wcol[(size_t)k4 * G4H];
                acc += (float)w.x * h_s[4 * k4 + 0]
                     + (float)w.y * h_s[4 * k4 + 1]
                     + (float)w.z * h_s[4 * k4 + 2]
                     + (float)w.w * h_s[4 * k4 + 3];
            }
            gates[j] = acc;
        }
        __syncthreads();

        if (tid < H_DIM) {
            float iv = gates[tid];
            float fv = gates[H_DIM + tid];
            float gv = gates[2 * H_DIM + tid];
            float ov = gates[3 * H_DIM + tid];
            float si = 1.0f / (1.0f + __expf(-iv));
            float sf = 1.0f / (1.0f + __expf(-fv));
            float so = 1.0f / (1.0f + __expf(-ov));
            float c  = sf * c_s[tid] + si * tanhf(gv);
            float h  = so * tanhf(c);
            c_s[tid] = c;
            h_s[tid] = h;
            tok[(size_t)t * D_IN + dir * H_DIM + tid] = h;
        }
        __syncthreads();
    }
}

// ---------------------------------------------------------------------------
// Column-parallel inclusive prefix sum (with leading zero row).
// ---------------------------------------------------------------------------
__global__ void prefix_kernel(const float* __restrict__ tok, float* __restrict__ pre) {
    int c = blockIdx.x * blockDim.x + threadIdx.x;
    if (c >= D_IN) return;
    float run = 0.0f;
    pre[c] = 0.0f;
    for (int t = 0; t < L_SEQ; ++t) {
        run += tok[(size_t)t * D_IN + c];
        pre[(size_t)(t + 1) * D_IN + c] = run;
    }
}

// Event embeddings: (prefix[en] - prefix[st]) / (en - st)
__global__ void event_kernel(const float* __restrict__ pre,
                             const long long* __restrict__ label_event,
                             float* __restrict__ ev) {
    int e = blockIdx.x;            // E blocks
    int d = threadIdx.x;           // 768 threads
    long long st = label_event[(size_t)e * 2];
    long long en = label_event[(size_t)e * 2 + 1];
    float inv = 1.0f / (float)(en - st);
    ev[(size_t)e * D_IN + d] =
        (pre[(size_t)en * D_IN + d] - pre[(size_t)st * D_IN + d]) * inv;
}

// Build bf16 feature A-matrix rows: [e1 | e2 | e1*e2]
__global__ void afeat_kernel(const float* __restrict__ ev,
                             const long long* __restrict__ pairs,
                             __bf16* __restrict__ Af) {
    int p = blockIdx.x;            // P blocks
    int d = threadIdx.x;           // 768 threads
    long long i1 = pairs[(size_t)p * 2];
    long long i2 = pairs[(size_t)p * 2 + 1];
    float a = ev[(size_t)i1 * D_IN + d];
    float b = ev[(size_t)i2 * D_IN + d];
    __bf16* row = Af + (size_t)p * KFEAT;
    row[d]        = (__bf16)a;
    row[768 + d]  = (__bf16)b;
    row[1536 + d] = (__bf16)(a * b);
}

// ---------------------------------------------------------------------------
// Head: scores = h2 @ W3^T + b3, log-softmax CE, block-reduced loss.
// out[0] = loss, out[1..] = scores (P,4)
// ---------------------------------------------------------------------------
__global__ void init_out_kernel(float* __restrict__ out) {
    if (threadIdx.x == 0 && blockIdx.x == 0) out[0] = 0.0f;
}

__global__ void head_kernel(const float* __restrict__ h2,
                            const float* __restrict__ w3,
                            const float* __restrict__ b3,
                            const long long* __restrict__ label,
                            float* __restrict__ out) {
    __shared__ float red[256];
    __shared__ float w3s[4 * 256];
    for (int i = threadIdx.x; i < 4 * 256; i += 256) w3s[i] = w3[i];
    __syncthreads();

    int p = blockIdx.x * 256 + threadIdx.x;
    float ce = 0.0f;
    if (p < P_NUM) {
        const float* hr = h2 + (size_t)p * 256;
        float s[4];
#pragma unroll
        for (int j = 0; j < 4; ++j) s[j] = b3[j];
        for (int k = 0; k < 256; ++k) {
            float hv = hr[k];
#pragma unroll
            for (int j = 0; j < 4; ++j) s[j] += w3s[j * 256 + k] * hv;
        }
        float mx  = fmaxf(fmaxf(s[0], s[1]), fmaxf(s[2], s[3]));
        float sum = expf(s[0] - mx) + expf(s[1] - mx) + expf(s[2] - mx) + expf(s[3] - mx);
        float lse = mx + logf(sum);
        int lbl = (int)label[p];
        ce = lse - s[lbl];                 // class weights are all 1.0
#pragma unroll
        for (int j = 0; j < 4; ++j) out[1 + (size_t)p * 4 + j] = s[j];
    }
    red[threadIdx.x] = ce;
    __syncthreads();
    for (int off = 128; off > 0; off >>= 1) {
        if (threadIdx.x < off) red[threadIdx.x] += red[threadIdx.x + off];
        __syncthreads();
    }
    if (threadIdx.x == 0) atomicAdd(&out[0], red[0]);
}

// ---------------------------------------------------------------------------
// Launcher
// ---------------------------------------------------------------------------
extern "C" void kernel_launch(void* const* d_in, const int* in_sizes, int n_in,
                              void* d_out, int out_size, void* d_ws, size_t ws_size,
                              hipStream_t stream) {
    const float* X    = (const float*)d_in[0];
    const float* wihf = (const float*)d_in[1];
    const float* whhf = (const float*)d_in[2];
    const float* bf_  = (const float*)d_in[3];
    const float* wihb = (const float*)d_in[4];
    const float* whhb = (const float*)d_in[5];
    const float* bb_  = (const float*)d_in[6];
    const float* w1   = (const float*)d_in[7];
    const float* b1   = (const float*)d_in[8];
    const float* w2   = (const float*)d_in[9];
    const float* b2   = (const float*)d_in[10];
    const float* w3   = (const float*)d_in[11];
    const float* b3   = (const float*)d_in[12];
    const long long* label_event = (const long long*)d_in[13];
    const long long* pairs       = (const long long*)d_in[14];
    const long long* label_tmp   = (const long long*)d_in[15];
    float* out = (float*)d_out;

    // workspace carve-up (256B aligned)
    size_t off = 0;
    auto alloc = [&](size_t bytes) {
        void* p = (char*)d_ws + off;
        off = (off + bytes + 255) & ~(size_t)255;
        return p;
    };
    __bf16* Xbf     = (__bf16*)alloc((size_t)L_SEQ * D_IN * 2);
    __bf16* WihBf   = (__bf16*)alloc((size_t)NGATE * D_IN * 2);
    float*  biasStk = (float*)alloc((size_t)NGATE * 4);
    __bf16* WhhT    = (__bf16*)alloc((size_t)2 * H_DIM * G4H * 2);
    __bf16* Wmlp1   = (__bf16*)alloc((size_t)D_IN * KFEAT * 2);
    __bf16* W2bf    = (__bf16*)alloc((size_t)256 * D_IN * 2);
    float*  Gx      = (float*)alloc((size_t)L_SEQ * NGATE * 4);
    float*  tok     = (float*)alloc((size_t)L_SEQ * D_IN * 4);
    float*  pre     = (float*)alloc((size_t)(L_SEQ + 1) * D_IN * 4);
    float*  ev      = (float*)alloc((size_t)E_NUM * D_IN * 4);
    __bf16* Afeat   = (__bf16*)alloc((size_t)P_NUM * KFEAT * 2);
    __bf16* h1bf    = (__bf16*)alloc((size_t)P_NUM * D_IN * 2);
    float*  h2      = (float*)alloc((size_t)P_NUM * 256 * 4);
    (void)ws_size;

    // ---- prep: bf16 conversions / folds ----
    k_f32_to_bf16<<<(L_SEQ * D_IN + 255) / 256, 256, 0, stream>>>(X, Xbf, L_SEQ * D_IN);
    k_build_wih<<<(NGATE * D_IN + 255) / 256, 256, 0, stream>>>(wihf, wihb, WihBf);
    k_build_bias<<<(NGATE + 255) / 256, 256, 0, stream>>>(bf_, bb_, biasStk);
    k_build_whhT<<<(2 * H_DIM * G4H + 255) / 256, 256, 0, stream>>>(whhf, whhb, WhhT);
    k_build_wmlp1<<<(D_IN * KFEAT + 255) / 256, 256, 0, stream>>>(w1, Wmlp1);
    k_f32_to_bf16<<<(256 * D_IN + 255) / 256, 256, 0, stream>>>(w2, W2bf, 256 * D_IN);

    // ---- GEMM 1: Gx = X @ Wih_stacked^T + bias  (M=4096, N=3072, K=768) ----
    {
        int tiles = (L_SEQ / 32) * (NGATE / 64);
        gemm_bf16_wmma<<<(tiles + 7) / 8, 256, 0, stream>>>(
            Xbf, D_IN, WihBf, D_IN, biasStk, Gx, nullptr,
            L_SEQ, NGATE, D_IN, 0);
    }

    // ---- LSTM recurrence: 2 blocks (fwd/bwd) ----
    lstm_kernel<<<2, 1024, 0, stream>>>(Gx, WhhT, tok);

    // ---- prefix sums / event embeddings / feature build ----
    prefix_kernel<<<(D_IN + 255) / 256, 256, 0, stream>>>(tok, pre);
    event_kernel<<<E_NUM, D_IN, 0, stream>>>(pre, label_event, ev);
    afeat_kernel<<<P_NUM, D_IN, 0, stream>>>(ev, pairs, Afeat);

    // ---- GEMM 2: h1 = relu(Afeat @ Wmlp1^T + b1)  (M=32768, N=768, K=2304) ----
    {
        int tiles = (P_NUM / 32) * (D_IN / 64);
        gemm_bf16_wmma<<<(tiles + 7) / 8, 256, 0, stream>>>(
            Afeat, KFEAT, Wmlp1, KFEAT, b1, nullptr, h1bf,
            P_NUM, D_IN, KFEAT, 1);
    }

    // ---- GEMM 3: h2 = relu(h1 @ W2^T + b2)  (M=32768, N=256, K=768) ----
    {
        int tiles = (P_NUM / 32) * (256 / 64);
        gemm_bf16_wmma<<<(tiles + 7) / 8, 256, 0, stream>>>(
            h1bf, D_IN, W2bf, D_IN, b2, h2, nullptr,
            P_NUM, 256, D_IN, 1);
    }

    // ---- head: scores + loss ----
    init_out_kernel<<<1, 1, 0, stream>>>(out);
    head_kernel<<<(P_NUM + 255) / 256, 256, 0, stream>>>(h2, w3, b3, label_tmp, out);
}
